// gnn_architecture_1_6322191859752
// MI455X (gfx1250) — matrix operand order repaired and checked by reading, not compiled, not run on hardware
//
#include <hip/hip_runtime.h>
#include <stddef.h>

typedef __attribute__((ext_vector_type(16))) __bf16 v16bf;
typedef __attribute__((ext_vector_type(8)))  __bf16 v8bf;
typedef __attribute__((ext_vector_type(4)))  __bf16 v4bf;
typedef __attribute__((ext_vector_type(8)))  float  v8f;

#define SLOPE 0.01f
#define EPS 1e-5f

// ---------------------------------------------------------------- helpers ---
__device__ __forceinline__ v16bf combine16(v8bf lo, v8bf hi) {
    return __builtin_shufflevector(lo, hi, 0,1,2,3,4,5,6,7,8,9,10,11,12,13,14,15);
}

// Load a 16-element bf16 fragment as two b128 chunks:
// uniform base (SGPR) + 32-bit lane offset (VGPR) -> SADDR-form global loads.
__device__ __forceinline__ v16bf load_frag(const char* ubase, unsigned off, int imm2nd) {
    v8bf lo = *(const v8bf*)(ubase + off);
    v8bf hi = *(const v8bf*)(ubase + off + imm2nd);
    return combine16(lo, hi);
}

// ------------------------------------------------------------ small kernels -
__global__ void deg_kernel(const int* __restrict__ col, float* __restrict__ deg, int E) {
    int e = blockIdx.x * blockDim.x + threadIdx.x;
    if (e < E) atomicAdd(&deg[col[e]], 1.0f);
}

__global__ void dis_kernel(float* __restrict__ deg, int N) {
    int i = blockIdx.x * blockDim.x + threadIdx.x;
    if (i < N) {
        float d = deg[i];
        deg[i] = (d > 0.0f) ? rsqrtf(d < 1.0f ? 1.0f : d) : 0.0f;
    }
}

__global__ void norm_kernel(const int* __restrict__ row, const int* __restrict__ col,
                            const float* __restrict__ dis, float* __restrict__ nrm, int E) {
    int e = blockIdx.x * blockDim.x + threadIdx.x;
    if (e < E) nrm[e] = dis[row[e]] * dis[col[e]];
}

// f32 -> bf16, 4 elements per thread (n multiple of 4: N*512)
__global__ void cvt_kernel(const float* __restrict__ src, __bf16* __restrict__ dst, size_t n4) {
    size_t i = (size_t)blockIdx.x * blockDim.x + threadIdx.x;
    if (i >= n4) return;
    float4 v = ((const float4*)src)[i];
    v4bf o = { (__bf16)v.x, (__bf16)v.y, (__bf16)v.z, (__bf16)v.w };
    ((v4bf*)dst)[i] = o;
}

// Wt[n][k] = bf16(W[k][n])   (W is fi x fo row-major)
__global__ void wt_kernel(const float* __restrict__ W, __bf16* __restrict__ Wt,
                          int K, int fo) {
    int i = blockIdx.x * blockDim.x + threadIdx.x;
    if (i >= fo * K) return;
    int n = i / K, k = i - n * K;
    Wt[i] = (__bf16)W[(size_t)k * fo + n];
}

// agg[col[e]] += norm[e] * x[row[e]]   (fi == 512 for all layers)
__global__ void scatter_kernel(const float* __restrict__ x, const int* __restrict__ row,
                               const int* __restrict__ col, const float* __restrict__ nrm,
                               float* __restrict__ agg) {
    int e = blockIdx.x;
    int r = row[e], c = col[e];
    float nv = nrm[e];
    const float4* xs = (const float4*)(x + (size_t)r * 512);   // 128 float4
    float* ad = agg + (size_t)c * 512;
    int f = threadIdx.x;                                       // 128 threads
    float4 v = xs[f];
    atomicAdd(&ad[4 * f + 0], v.x * nv);
    atomicAdd(&ad[4 * f + 1], v.y * nv);
    atomicAdd(&ad[4 * f + 2], v.z * nv);
    atomicAdd(&ad[4 * f + 3], v.w * nv);
}

// ------------------------------------------------------------------- GEMM ---
// out[M,fo] = Xb[M,K] @ W0 + Ab[M,K] @ W1, weights pre-transposed bf16 [fo,K].
// Block: 256 threads = 8 waves (4 along M x 2 along N); block tile 128x128.
// Wave tile 32x64 = 2x4 WMMA frags -> 8 WMMA per 12 b128 loads per k-step.
// Addressing: uniform base advances with k (SALU only); per-lane parts are
// loop-invariant u32 offsets -> SADDR-form loads, no VALU in the k-loop,
// no WMMA->VALU hazard NOPs.
__global__ __launch_bounds__(256)
void gemm_kernel(const __bf16* __restrict__ Xb, const __bf16* __restrict__ Ab,
                 const __bf16* __restrict__ W0t, const __bf16* __restrict__ W1t,
                 float* __restrict__ out, int M, int K, int fo) {
    const int lane = threadIdx.x & 31;
    const int wave = threadIdx.x >> 5;
    const int l16  = lane & 15;
    const int h    = lane >> 4;          // lane half (wave32)
    const int wm   = (wave & 3) * 32;    // 4 waves along M
    const int wn   = (wave >> 2) * 64;   // 2 waves along N
    const int bm   = blockIdx.x * 128;
    const int bn   = blockIdx.y * 128;

    int r0 = bm + wm + l16;
    int r1 = r0 + 16;
    int r0c = r0 < M ? r0 : M - 1;       // clamp (stores are guarded)
    int r1c = r1 < M ? r1 : M - 1;

    // loop-invariant 32-bit byte offsets (per lane)
    const unsigned oa0 = (unsigned)(r0c * K + h * 8) * 2u;
    const unsigned oa1 = (unsigned)(r1c * K + h * 8) * 2u;
    unsigned ob[4];
    #pragma unroll
    for (int j = 0; j < 4; ++j)
        ob[j] = (unsigned)((bn + wn + j * 16 + l16) * K + h * 8) * 2u;

    v8f acc[2][4];
    #pragma unroll
    for (int i = 0; i < 2; ++i)
        #pragma unroll
        for (int j = 0; j < 4; ++j)
            acc[i][j] = (v8f){};

    #pragma unroll
    for (int pass = 0; pass < 2; ++pass) {
        const char* Abase = (const char*)(pass ? Ab  : Xb);   // uniform
        const char* Wbase = (const char*)(pass ? W1t : W0t);  // uniform
        #pragma unroll 4
        for (int k = 0; k < K; k += 32) {
            const char* ua = Abase + (size_t)(2 * k);         // uniform, SALU advance
            const char* uw = Wbase + (size_t)(2 * k);
            // A frag: row m, K in [8h,8h+8) U [16+8h,16+8h+8)  -> chunks 32B apart
            v16bf a0 = load_frag(ua, oa0, 32);
            v16bf a1 = load_frag(ua, oa1, 32);
            // B frag: col n, K in [8h,8h+8) U [16+8h,16+8h+8)  -> chunks 32B apart
            v16bf b0 = load_frag(uw, ob[0], 32);
            v16bf b1 = load_frag(uw, ob[1], 32);
            v16bf b2 = load_frag(uw, ob[2], 32);
            v16bf b3 = load_frag(uw, ob[3], 32);
            acc[0][0] = __builtin_amdgcn_wmma_f32_16x16x32_bf16(false, a0, false, b0, (short)0, acc[0][0], false, false);
            acc[1][0] = __builtin_amdgcn_wmma_f32_16x16x32_bf16(false, a1, false, b0, (short)0, acc[1][0], false, false);
            acc[0][1] = __builtin_amdgcn_wmma_f32_16x16x32_bf16(false, a0, false, b1, (short)0, acc[0][1], false, false);
            acc[1][1] = __builtin_amdgcn_wmma_f32_16x16x32_bf16(false, a1, false, b1, (short)0, acc[1][1], false, false);
            acc[0][2] = __builtin_amdgcn_wmma_f32_16x16x32_bf16(false, a0, false, b2, (short)0, acc[0][2], false, false);
            acc[1][2] = __builtin_amdgcn_wmma_f32_16x16x32_bf16(false, a1, false, b2, (short)0, acc[1][2], false, false);
            acc[0][3] = __builtin_amdgcn_wmma_f32_16x16x32_bf16(false, a0, false, b3, (short)0, acc[0][3], false, false);
            acc[1][3] = __builtin_amdgcn_wmma_f32_16x16x32_bf16(false, a1, false, b3, (short)0, acc[1][3], false, false);
        }
    }

    // C/D layout: VGPR v -> row (8h + v), col l16
    #pragma unroll
    for (int v = 0; v < 8; ++v) {
        int rr0 = bm + wm + 8 * h + v;
        int rr1 = rr0 + 16;
        if (rr0 < M) {
            float* o = out + (size_t)rr0 * fo + bn + wn + l16;
            #pragma unroll
            for (int j = 0; j < 4; ++j) o[j * 16] = acc[0][j][v];
        }
        if (rr1 < M) {
            float* o = out + (size_t)rr1 * fo + bn + wn + l16;
            #pragma unroll
            for (int j = 0; j < 4; ++j) o[j * 16] = acc[1][j][v];
        }
    }
}

// --------------------------------------------------------------- BN + act ---
__global__ void bnstat_kernel(const float* __restrict__ out, float* __restrict__ sum,
                              float* __restrict__ sumsq, int M, int fo) {
    int c = blockIdx.y * blockDim.x + threadIdx.x;
    if (c >= fo) return;
    int r0 = blockIdx.x * 1024;
    int r1 = r0 + 1024 < M ? r0 + 1024 : M;
    float s = 0.0f, s2 = 0.0f;
    for (int r = r0; r < r1; ++r) {
        float v = out[(size_t)r * fo + c];
        s += v; s2 += v * v;
    }
    atomicAdd(&sum[c], s);
    atomicAdd(&sumsq[c], s2);
}

__global__ void bnfinal_kernel(const float* __restrict__ sum, const float* __restrict__ sumsq,
                               const float* __restrict__ gamma, const float* __restrict__ beta,
                               float* __restrict__ scale, float* __restrict__ shift,
                               int fo, float invM) {
    int c = blockIdx.x * blockDim.x + threadIdx.x;
    if (c >= fo) return;
    float mean = sum[c] * invM;
    float var  = sumsq[c] * invM - mean * mean;
    float inv  = rsqrtf(var + EPS);
    float sc   = gamma[c] * inv;
    scale[c] = sc;
    shift[c] = beta[c] - mean * sc;
}

// y = LeakyReLU(out*scale + shift); 4 elems/thread; writes f32 (+ optional bf16)
__global__ void act_kernel(const float* __restrict__ src, const float* __restrict__ scale,
                           const float* __restrict__ shift, float* __restrict__ dstf,
                           __bf16* __restrict__ dstb, size_t n4, int fomask) {
    size_t i = (size_t)blockIdx.x * blockDim.x + threadIdx.x;
    if (i >= n4) return;
    int c = (int)((4 * i) & (size_t)fomask);   // fo is a power of two; no wrap in 4
    float4 v = ((const float4*)src)[i];
    float4 r;
    r.x = v.x * scale[c + 0] + shift[c + 0];
    r.y = v.y * scale[c + 1] + shift[c + 1];
    r.z = v.z * scale[c + 2] + shift[c + 2];
    r.w = v.w * scale[c + 3] + shift[c + 3];
    r.x = r.x > 0.0f ? r.x : SLOPE * r.x;
    r.y = r.y > 0.0f ? r.y : SLOPE * r.y;
    r.z = r.z > 0.0f ? r.z : SLOPE * r.z;
    r.w = r.w > 0.0f ? r.w : SLOPE * r.w;
    ((float4*)dstf)[i] = r;
    if (dstb) {
        v4bf o = { (__bf16)r.x, (__bf16)r.y, (__bf16)r.z, (__bf16)r.w };
        ((v4bf*)dstb)[i] = o;
    }
}

// ------------------------------------------------------------------ launch --
extern "C" void kernel_launch(void* const* d_in, const int* in_sizes, int n_in,
                              void* d_out, int out_size, void* d_ws, size_t ws_size,
                              hipStream_t stream) {
    const float* y  = (const float*)d_in[0];
    const int*   ei = (const int*)d_in[1];
    const int N = in_sizes[0] / 512;
    const int E = in_sizes[1] / 2;
    const int* row = ei;
    const int* col = ei + E;

    const float* w0g[3]   = {(const float*)d_in[2],  (const float*)d_in[7],  (const float*)d_in[12]};
    const float* w1g[3]   = {(const float*)d_in[3],  (const float*)d_in[8],  (const float*)d_in[13]};
    const float* gamma[3] = {(const float*)d_in[5],  (const float*)d_in[10], (const float*)d_in[14]};
    const float* beta[3]  = {(const float*)d_in[6],  (const float*)d_in[11], (const float*)d_in[15]};
    const int fo3[3] = {512, 512, 256};
    // bias (d_in[4], d_in[9]) cancels inside BatchNorm -> unused (exact)

    char* ws = (char*)d_ws;
    size_t off = 0;
    auto carve = [&](size_t bytes) -> void* {
        void* p = ws + off;
        off += (bytes + 255) & ~(size_t)255;
        return p;
    };
    float*  bufA   = (float*)carve((size_t)N * 512 * 4);
    float*  bufB   = (float*)carve((size_t)N * 512 * 4);
    float*  aggf   = (float*)carve((size_t)N * 512 * 4);
    __bf16* xb     = (__bf16*)carve((size_t)N * 512 * 2);
    __bf16* aggb   = (__bf16*)carve((size_t)N * 512 * 2);
    __bf16* w0t    = (__bf16*)carve(512 * 512 * 2);
    __bf16* w1t    = (__bf16*)carve(512 * 512 * 2);
    float*  nrm    = (float*)carve((size_t)E * 4);
    float*  dis    = (float*)carve((size_t)N * 4);
    float*  csum   = (float*)carve(512 * 4);
    float*  csumsq = (float*)carve(512 * 4);
    float*  bscale = (float*)carve(512 * 4);
    float*  bshift = (float*)carve(512 * 4);

    // GCN normalization (once)
    hipMemsetAsync(dis, 0, (size_t)N * 4, stream);
    deg_kernel<<<(E + 255) / 256, 256, 0, stream>>>(col, dis, E);
    dis_kernel<<<(N + 255) / 256, 256, 0, stream>>>(dis, N);
    norm_kernel<<<(E + 255) / 256, 256, 0, stream>>>(row, col, dis, nrm, E);

    // bf16 copy of layer-0 input
    {
        size_t n4 = (size_t)N * 512 / 4;
        cvt_kernel<<<(int)((n4 + 255) / 256), 256, 0, stream>>>(y, xb, n4);
    }

    const float* xf = y;
    float* outs[3] = {bufA, bufB, bufA};
    float* actf[3] = {bufA, bufB, (float*)d_out};

    for (int L = 0; L < 3; ++L) {
        const int F = fo3[L];
        // scatter aggregation
        hipMemsetAsync(aggf, 0, (size_t)N * 512 * 4, stream);
        scatter_kernel<<<E, 128, 0, stream>>>(xf, row, col, nrm, aggf);
        size_t n4 = (size_t)N * 512 / 4;
        cvt_kernel<<<(int)((n4 + 255) / 256), 256, 0, stream>>>(aggf, aggb, n4);
        // weight transpose + bf16
        int wcount = F * 512;
        wt_kernel<<<(wcount + 255) / 256, 256, 0, stream>>>(w0g[L], w0t, 512, F);
        wt_kernel<<<(wcount + 255) / 256, 256, 0, stream>>>(w1g[L], w1t, 512, F);
        // fused dual-GEMM (WMMA bf16, f32 accumulate)
        dim3 gg((N + 127) / 128, F / 128);
        gemm_kernel<<<gg, 256, 0, stream>>>(xb, aggb, w0t, w1t, outs[L], N, 512, F);
        // batch-norm statistics
        hipMemsetAsync(csum, 0, F * 4, stream);
        hipMemsetAsync(csumsq, 0, F * 4, stream);
        dim3 gs((N + 1023) / 1024, (F + 255) / 256);
        bnstat_kernel<<<gs, 256, 0, stream>>>(outs[L], csum, csumsq, N, F);
        bnfinal_kernel<<<(F + 255) / 256, 256, 0, stream>>>(csum, csumsq, gamma[L], beta[L],
                                                            bscale, bshift, F, 1.0f / (float)N);
        // BN + LeakyReLU (+ bf16 copy for next layer's GEMM)
        size_t no4 = (size_t)N * F / 4;
        __bf16* dstb = (L < 2) ? xb : nullptr;
        act_kernel<<<(int)((no4 + 255) / 256), 256, 0, stream>>>(outs[L], bscale, bshift,
                                                                 actf[L], dstb, no4, F - 1);
        xf = actf[L];
    }
}
